// EnsembleDNN_88862873355052
// MI455X (gfx1250) — matrix operand order repaired and checked by reading, not compiled, not run on hardware
//
#include <hip/hip_runtime.h>

// ---------------------------------------------------------------------------
// EnsembleDNN on MI455X (gfx1250):
//   out[b, 0, e] = W3[e]·relu(W2[e]^T·relu(W1[e]^T·x[b] + b1) + b2) + b3
// Fused per-(expert, 128-row) workgroup. bf16 WMMA (f32 accumulate), all
// intermediates in the 320KB WGP LDS. W1/W2 are streamed HBM->LDS by the
// Tensor Data Mover (double-buffered, TENSORcnt-tracked), converted once to
// bf16 in LDS, and fed to v_wmma_f32_16x16x32_bf16. Layer-1 AI = 512 flop/B
// -> compute-bound, so the bf16 matrix path is the right precision.
// ---------------------------------------------------------------------------

#define E_   100
#define D_   2048
#define H1_  512
#define H2_  256
#define B_   1024
#define BT_  128          // rows per workgroup
#define NC_  128          // N-panel width per wave per pass
#define KP_  40           // padded K-stride (bf16 elems) of staging tiles
#define H1P_ 520          // padded row stride of h1 tile (bf16 elems)
#define H2P_ 264          // padded row stride of h2 tile (bf16 elems)

typedef __attribute__((ext_vector_type(16))) __bf16        v16bf;
typedef __attribute__((ext_vector_type(8)))  float         v8f;
typedef __attribute__((ext_vector_type(4)))  float         v4f;
typedef __attribute__((ext_vector_type(4)))  unsigned int  v4u;
typedef __attribute__((ext_vector_type(2)))  unsigned int  v2u;
typedef __attribute__((ext_vector_type(8)))  int           v8i;
typedef __attribute__((ext_vector_type(4)))  int           v4i;

union Frag { v16bf v; v4u q[2]; };

#if __has_builtin(__builtin_amdgcn_tensor_load_to_lds)
#define USE_TDM 1
#else
#define USE_TDM 0
#endif

// f32 -> bf16 via native conversion (lowers to v_cvt_pk_bf16_f32 on gfx1250)
static __device__ inline unsigned short f2bf(float x) {
    __bf16 h = (__bf16)x;
    return __builtin_bit_cast(unsigned short, h);
}
static __device__ inline float bf2f(unsigned short h) {
    return __uint_as_float(((unsigned)h) << 16);
}

// LDS byte offsets (all 16B aligned)
#define OFF_H1   0u        // 128*520*2 = 133120
#define OFF_H2   133120u   // 128*264*2 =  67584
#define OFF_SX   200704u   // 128*40*2  =  10240
#define OFF_SW   210944u   // 128*40*2  =  10240
#define OFF_B1   221184u   // 512*4
#define OFF_B2   223232u   // 256*4
#define OFF_W3   224256u   // 256*4
#define OFF_B3   225280u   // 16
#define OFF_RAW  225296u   // 2 x (32*128*4) = 32768 raw f32 TDM buffers
#define RAW_BYTES 16384u
#define SMEM_BYTES 258064u

#if USE_TDM
// Issue a TDM 2D tile load: tile_y rows x tile_x f32 elements, row stride
// `stride` elements, from global `gaddr` into LDS byte offset `lds_off`.
// D# layout per CDNA5 ISA ch.8 (group0: count/lds/global/type,
// group1: data_size, dims, tile dims, stride). Groups 2/3 zero (2D tensor).
// This toolchain's builtin takes 6 args (g0, g1, g2, g3, g4, cpol).
static __device__ inline void tdm_load_2d(unsigned lds_off, const void* gaddr,
                                          unsigned tile_x, unsigned tile_y,
                                          unsigned stride) {
    unsigned long long ga = (unsigned long long)gaddr;
    const unsigned td0 = 1u << 20, td1 = 1u << 20;   // large: no OOB clipping
    v4u g0;
    g0.x = 1u;                                      // count=1 valid descriptor
    g0.y = lds_off;                                 // LDS byte address
    g0.z = (unsigned)(ga & 0xFFFFFFFFu);            // global_addr[31:0]
    g0.w = (unsigned)((ga >> 32) & 0x01FFFFFFu)     // global_addr[56:32]
         | (2u << 30);                              // type=2 ("image")
    v8i g1;
    g1[0] = (int)(2u << 16);                        // wg_mask=0, data_size=4B
    g1[1] = (int)((td0 & 0xFFFFu) << 16);           // tensor_dim0 lo16
    g1[2] = (int)(((td0 >> 16) & 0xFFFFu) | ((td1 & 0xFFFFu) << 16));
    g1[3] = (int)(((td1 >> 16) & 0xFFFFu) | ((tile_x & 0xFFFFu) << 16));
    g1[4] = (int)(tile_y & 0xFFFFu);                // tile_dim1 (tile_dim2=0)
    g1[5] = (int)stride;                            // tensor_dim0_stride lo32
    g1[6] = 0;                                      // stride hi16, dim1_stride lo
    g1[7] = 0;
    v4i z4 = {0, 0, 0, 0};
    v8i z8 = {0, 0, 0, 0, 0, 0, 0, 0};
    __builtin_amdgcn_tensor_load_to_lds(g0, g1, z4, z4, z8, 0);
}
#endif

__global__ void __launch_bounds__(256)
ensemble_mlp_kernel(const float* __restrict__ x,  const float* __restrict__ W1,
                    const float* __restrict__ b1, const float* __restrict__ W2,
                    const float* __restrict__ b2, const float* __restrict__ W3,
                    const float* __restrict__ b3, float* __restrict__ out)
{
    extern __shared__ char smem[];
    unsigned short* h1s = (unsigned short*)(smem + OFF_H1);
    unsigned short* h2s = (unsigned short*)(smem + OFF_H2);
    unsigned short* sX  = (unsigned short*)(smem + OFF_SX);
    unsigned short* sW  = (unsigned short*)(smem + OFF_SW);
    float* b1s = (float*)(smem + OFF_B1);
    float* b2s = (float*)(smem + OFF_B2);
    float* w3s = (float*)(smem + OFF_W3);
    float* b3s = (float*)(smem + OFF_B3);
    float* raw0 = (float*)(smem + OFF_RAW);
    float* raw1 = (float*)(smem + OFF_RAW + RAW_BYTES);

    const int tid  = threadIdx.x;
    const int lane = tid & 31;
    const int wave = tid >> 5;
    const int e    = blockIdx.y;
    const int b0   = blockIdx.x * BT_;

    const int m0   = wave * 16;             // this wave's M strip
    const int nl   = lane & 15;             // N within a 16-wide tile
    const int ml   = m0 + nl;               // this lane's A-matrix row
    const int kb0  = (lane < 16) ? 0 : 8;   // A fragment K sub-base (16-bit layout)
    const int kbB  = (lane < 16) ? 0 : 8;   // B fragment K base (16-bit layout)
    const int rOff = (lane < 16) ? 0 : 8;   // C/D row offset for upper half-wave

    // ---- preload per-expert biases and W3 column into LDS ----
    for (int i = tid; i < H1_; i += 256) b1s[i] = b1[(size_t)e * H1_ + i];
    for (int i = tid; i < H2_; i += 256) {
        b2s[i] = b2[(size_t)e * H2_ + i];
        w3s[i] = W3[(size_t)e * H2_ + i];     // O == 1
    }
    if (tid == 0) b3s[0] = b3[e];

    const float* xBase  = x  + (size_t)b0 * D_;
    const float* w1Base = W1 + (size_t)e * D_ * H1_;
    const float* w2Base = W2 + (size_t)e * H1_ * H2_;

    // =============== Phase 1: h1 = relu(x * W1[e] + b1[e]) ===============
    for (int nc = 0; nc < H1_; nc += NC_) {
        v8f acc[8] = {};
#if USE_TDM
        if (wave == 0)   // prime the pipeline: chunk k=0 -> raw0
            tdm_load_2d(OFF_RAW, w1Base + nc, NC_, 32, H1_);
#endif
        for (int k = 0, it = 0; k < D_; k += 32, ++it) {
            const int cur = it & 1;
            const float* rawCur = cur ? raw1 : raw0;
#if USE_TDM
            if (wave == 0) {
                if (k + 32 < D_) {   // DMA next chunk into the other buffer
                    tdm_load_2d(OFF_RAW + (unsigned)(cur ^ 1) * RAW_BYTES,
                                w1Base + (size_t)(k + 32) * H1_ + nc, NC_, 32, H1_);
                    __builtin_amdgcn_s_wait_tensorcnt(1);  // chunk k complete
                } else {
                    __builtin_amdgcn_s_wait_tensorcnt(0);
                }
            }
#endif
            __syncthreads();   // raw chunk visible; prev-iter LDS reads retired
            // stage x[.,k..k+32) as bf16 [128][32] (f32x4 loads, b64 packed stores)
            for (int idx = tid; idx < BT_ * 8; idx += 256) {
                int m = idx >> 3, k4 = (idx & 7) * 4;
                v4f xv = *(const v4f*)(xBase + (size_t)m * D_ + k + k4);
                v2u pk;
                pk.x = (unsigned)f2bf(xv.x) | ((unsigned)f2bf(xv.y) << 16);
                pk.y = (unsigned)f2bf(xv.z) | ((unsigned)f2bf(xv.w) << 16);
                *(v2u*)(sX + m * KP_ + k4) = pk;
            }
            // convert raw W1 chunk [32][128] f32 -> transposed bf16 [N=128][K=32]
            for (int idx = tid; idx < 32 * 32; idx += 256) {
                int kk = idx >> 5, n4 = (idx & 31) * 4;
#if USE_TDM
                v4f wv = *(const v4f*)(rawCur + kk * NC_ + n4);
#else
                v4f wv = *(const v4f*)(w1Base + (size_t)(k + kk) * H1_ + nc + n4);
#endif
                sW[(n4 + 0) * KP_ + kk] = f2bf(wv.x);
                sW[(n4 + 1) * KP_ + kk] = f2bf(wv.y);
                sW[(n4 + 2) * KP_ + kk] = f2bf(wv.z);
                sW[(n4 + 3) * KP_ + kk] = f2bf(wv.w);
            }
            __syncthreads();

            // load A fragment + all 8 B fragments, then burst the WMMAs
            Frag a;
            const unsigned short* ar = sX + ml * KP_;
            a.q[0] = *(const v4u*)(ar + kb0);
            a.q[1] = *(const v4u*)(ar + kb0 + 16);
            Frag bm[8];
#pragma unroll
            for (int nt = 0; nt < 8; ++nt) {
                const unsigned short* br = sW + (nt * 16 + nl) * KP_ + kbB;
                bm[nt].q[0] = ((const v4u*)br)[0];
                bm[nt].q[1] = ((const v4u*)br)[2];
            }
#pragma unroll
            for (int nt = 0; nt < 8; ++nt)
                acc[nt] = __builtin_amdgcn_wmma_f32_16x16x32_bf16(
                    false, a.v, false, bm[nt].v, (short)0, acc[nt], false, false);
        }
        // epilogue: bias + relu -> h1 tile (bf16) in LDS
#pragma unroll
        for (int nt = 0; nt < 8; ++nt) {
            int col = nc + nt * 16 + nl;
            float bias = b1s[col];
#pragma unroll
            for (int r = 0; r < 8; ++r) {
                int row = m0 + r + rOff;
                float v = acc[nt][r] + bias;
                v = v > 0.f ? v : 0.f;
                h1s[row * H1P_ + col] = f2bf(v);
            }
        }
    }

    // =============== Phase 2: h2 = relu(h1 * W2[e] + b2[e]) ===============
    for (int nc = 0; nc < H2_; nc += NC_) {
        v8f acc[8] = {};
#if USE_TDM
        if (wave == 0)
            tdm_load_2d(OFF_RAW, w2Base + nc, NC_, 32, H2_);
#endif
        for (int k = 0, it = 0; k < H1_; k += 32, ++it) {
            const int cur = it & 1;
            const float* rawCur = cur ? raw1 : raw0;
#if USE_TDM
            if (wave == 0) {
                if (k + 32 < H1_) {
                    tdm_load_2d(OFF_RAW + (unsigned)(cur ^ 1) * RAW_BYTES,
                                w2Base + (size_t)(k + 32) * H2_ + nc, NC_, 32, H2_);
                    __builtin_amdgcn_s_wait_tensorcnt(1);
                } else {
                    __builtin_amdgcn_s_wait_tensorcnt(0);
                }
            }
#endif
            __syncthreads();
            // convert raw W2 chunk [32][128] f32 -> transposed bf16 [N=128][K=32]
            for (int idx = tid; idx < 32 * 32; idx += 256) {
                int kk = idx >> 5, n4 = (idx & 31) * 4;
#if USE_TDM
                v4f wv = *(const v4f*)(rawCur + kk * NC_ + n4);
#else
                v4f wv = *(const v4f*)(w2Base + (size_t)(k + kk) * H2_ + nc + n4);
#endif
                sW[(n4 + 0) * KP_ + kk] = f2bf(wv.x);
                sW[(n4 + 1) * KP_ + kk] = f2bf(wv.y);
                sW[(n4 + 2) * KP_ + kk] = f2bf(wv.z);
                sW[(n4 + 3) * KP_ + kk] = f2bf(wv.w);
            }
            __syncthreads();

            Frag a;  // A comes straight from the bf16 h1 tile in LDS
            const unsigned short* ar = h1s + ml * H1P_ + k;
            a.q[0] = *(const v4u*)(ar + kb0);
            a.q[1] = *(const v4u*)(ar + kb0 + 16);
            Frag bm[8];
#pragma unroll
            for (int nt = 0; nt < 8; ++nt) {
                const unsigned short* br = sW + (nt * 16 + nl) * KP_ + kbB;
                bm[nt].q[0] = ((const v4u*)br)[0];
                bm[nt].q[1] = ((const v4u*)br)[2];
            }
#pragma unroll
            for (int nt = 0; nt < 8; ++nt)
                acc[nt] = __builtin_amdgcn_wmma_f32_16x16x32_bf16(
                    false, a.v, false, bm[nt].v, (short)0, acc[nt], false, false);
        }
#pragma unroll
        for (int nt = 0; nt < 8; ++nt) {
            int col = nc + nt * 16 + nl;
            float bias = b2s[col];
#pragma unroll
            for (int r = 0; r < 8; ++r) {
                int row = m0 + r + rOff;
                float v = acc[nt][r] + bias;
                v = v > 0.f ? v : 0.f;
                h2s[row * H2P_ + col] = f2bf(v);
            }
        }
    }
    __syncthreads();

    // =============== Phase 3: out[b,0,e] = h2 . W3[e] + b3[e] ===============
    if (tid < BT_) {
        const unsigned short* hr = h2s + tid * H2P_;
        float sum = b3s[0];
#pragma unroll 4
        for (int kk = 0; kk < H2_; ++kk)
            sum += bf2f(hr[kk]) * w3s[kk];
        out[(size_t)(b0 + tid) * E_ + e] = sum;   // [B, O=1, E] layout
    }
}

extern "C" void kernel_launch(void* const* d_in, const int* in_sizes, int n_in,
                              void* d_out, int out_size, void* d_ws, size_t ws_size,
                              hipStream_t stream) {
    (void)in_sizes; (void)n_in; (void)d_ws; (void)ws_size; (void)out_size;
    const float* x  = (const float*)d_in[0];
    const float* W1 = (const float*)d_in[1];
    const float* b1 = (const float*)d_in[2];
    const float* W2 = (const float*)d_in[3];
    const float* b2 = (const float*)d_in[4];
    const float* W3 = (const float*)d_in[5];
    const float* b3 = (const float*)d_in[6];
    float* out = (float*)d_out;

    // CDNA5 allows up to 320KB LDS per workgroup; opt in to >64KB dynamic LDS.
    (void)hipFuncSetAttribute((const void*)ensemble_mlp_kernel,
                              hipFuncAttributeMaxDynamicSharedMemorySize,
                              (int)SMEM_BYTES);

    dim3 grid(B_ / BT_, E_, 1);   // 8 B-tiles x 100 experts
    ensemble_mlp_kernel<<<grid, 256, SMEM_BYTES, stream>>>(
        x, W1, b1, W2, b2, W3, b3, out);
}
